// QABM_48679159333341
// MI455X (gfx1250) — matrix operand order repaired and checked by reading, not compiled, not run on hardware
//
#include <hip/hip_runtime.h>
#include <hip/hip_bf16.h>

typedef __attribute__((ext_vector_type(16))) __bf16 v16bf;
typedef __attribute__((ext_vector_type(8)))  float  v8f;

__device__ __forceinline__ unsigned short f2bf(float x) {
  unsigned u = __float_as_uint(x);
  unsigned r = u + 0x7FFFu + ((u >> 16) & 1u);   // round-to-nearest-even
  return (unsigned short)(r >> 16);
}

union BfPack { v16bf v; unsigned short u[16]; };

// ---------------------------------------------------------------------------
// Pass A: per-batch precompute. One block per batch, 256 threads.
//  - K = wt@Wk, V = wt@Wv              (20x256 each, LDS)
//  - M[c2][l] = sum_o Wq[o][c2]*K[l][o] * (1/16)   (fold Wq into keys, pre-scale)
//  - blob: M packed to bf16 in exact WMMA B-operand layout (8 K-steps x 2 N-tiles)
//  - G = V V^T (20x20 Gram)  ;  1+gamma, beta, normalized vec2, mask bits
//  - zero the atomic-max slots
// ---------------------------------------------------------------------------
__global__ void precompute_kernel(
    const float* __restrict__ wt, const float* __restrict__ sv,
    const int* __restrict__ mask,
    const float* __restrict__ Wq, const float* __restrict__ Wk, const float* __restrict__ Wv,
    const float* __restrict__ gw1, const float* __restrict__ gb1,
    const float* __restrict__ gw2, const float* __restrict__ gb2,
    const float* __restrict__ bw1, const float* __restrict__ bb1,
    const float* __restrict__ bw2, const float* __restrict__ bb2,
    const float* __restrict__ dw1, const float* __restrict__ db1,
    const float* __restrict__ dw2, const float* __restrict__ db2,
    unsigned short* __restrict__ blob, float* __restrict__ Gm,
    float* __restrict__ g1p, float* __restrict__ beta_o,
    float* __restrict__ vec2_o, unsigned* __restrict__ mbits,
    float* __restrict__ maxg)
{
  __shared__ float s_sent[768];
  __shared__ float s_h[256];
  __shared__ float s_K[5120];
  __shared__ float s_V[5120];
  __shared__ float s_M[5120];
  const int b = blockIdx.x;
  const int t = threadIdx.x;

  for (int i = t; i < 768; i += 256) s_sent[i] = sv[b * 768 + i];
  if (t < 4) maxg[t * 16 + b] = 0.0f;                 // zero bits == 0.0f for uint atomicMax
  if (t == 0) {
    unsigned m = 0;
    for (int l = 0; l < 20; ++l) if (mask[b * 20 + l] != 0) m |= (1u << l);
    mbits[b] = m;
  }
  __syncthreads();

  // ---- gamma MLP ----
  {
    float a0 = 0, a1 = 0, a2 = 0, a3 = 0;
    for (int i = 0; i < 768; i += 4) {
      a0 = fmaf(s_sent[i + 0], gw1[(size_t)(i + 0) * 256 + t], a0);
      a1 = fmaf(s_sent[i + 1], gw1[(size_t)(i + 1) * 256 + t], a1);
      a2 = fmaf(s_sent[i + 2], gw1[(size_t)(i + 2) * 256 + t], a2);
      a3 = fmaf(s_sent[i + 3], gw1[(size_t)(i + 3) * 256 + t], a3);
    }
    s_h[t] = fmaxf(a0 + a1 + a2 + a3 + gb1[t], 0.0f);
  }
  __syncthreads();
  {
    float a0 = 0, a1 = 0, a2 = 0, a3 = 0;
    for (int j = 0; j < 256; j += 4) {
      a0 = fmaf(s_h[j + 0], gw2[(size_t)(j + 0) * 256 + t], a0);
      a1 = fmaf(s_h[j + 1], gw2[(size_t)(j + 1) * 256 + t], a1);
      a2 = fmaf(s_h[j + 2], gw2[(size_t)(j + 2) * 256 + t], a2);
      a3 = fmaf(s_h[j + 3], gw2[(size_t)(j + 3) * 256 + t], a3);
    }
    g1p[b * 256 + t] = 1.0f + (a0 + a1 + a2 + a3 + gb2[t]);
  }
  __syncthreads();

  // ---- beta MLP ----
  {
    float a0 = 0, a1 = 0, a2 = 0, a3 = 0;
    for (int i = 0; i < 768; i += 4) {
      a0 = fmaf(s_sent[i + 0], bw1[(size_t)(i + 0) * 256 + t], a0);
      a1 = fmaf(s_sent[i + 1], bw1[(size_t)(i + 1) * 256 + t], a1);
      a2 = fmaf(s_sent[i + 2], bw1[(size_t)(i + 2) * 256 + t], a2);
      a3 = fmaf(s_sent[i + 3], bw1[(size_t)(i + 3) * 256 + t], a3);
    }
    s_h[t] = fmaxf(a0 + a1 + a2 + a3 + bb1[t], 0.0f);
  }
  __syncthreads();
  {
    float a0 = 0, a1 = 0, a2 = 0, a3 = 0;
    for (int j = 0; j < 256; j += 4) {
      a0 = fmaf(s_h[j + 0], bw2[(size_t)(j + 0) * 256 + t], a0);
      a1 = fmaf(s_h[j + 1], bw2[(size_t)(j + 1) * 256 + t], a1);
      a2 = fmaf(s_h[j + 2], bw2[(size_t)(j + 2) * 256 + t], a2);
      a3 = fmaf(s_h[j + 3], bw2[(size_t)(j + 3) * 256 + t], a3);
    }
    beta_o[b * 256 + t] = a0 + a1 + a2 + a3 + bb2[t];
  }
  __syncthreads();

  // ---- vec2 MLP (768 -> 64 -> 2, normalize) ----
  if (t < 64) {
    float a0 = 0, a1 = 0, a2 = 0, a3 = 0;
    for (int i = 0; i < 768; i += 4) {
      a0 = fmaf(s_sent[i + 0], dw1[(size_t)(i + 0) * 64 + t], a0);
      a1 = fmaf(s_sent[i + 1], dw1[(size_t)(i + 1) * 64 + t], a1);
      a2 = fmaf(s_sent[i + 2], dw1[(size_t)(i + 2) * 64 + t], a2);
      a3 = fmaf(s_sent[i + 3], dw1[(size_t)(i + 3) * 64 + t], a3);
    }
    s_h[t] = fmaxf(a0 + a1 + a2 + a3 + db1[t], 0.0f);
  }
  __syncthreads();
  if (t < 2) {
    float a = db2[t];
    for (int j = 0; j < 64; ++j) a = fmaf(s_h[j], dw2[(size_t)j * 2 + t], a);
    s_h[128 + t] = a;
  }
  __syncthreads();
  if (t == 0) {
    float vx = s_h[128], vy = s_h[129];
    float nrm = fmaxf(sqrtf(vx * vx + vy * vy), 1e-12f);
    vec2_o[b * 2 + 0] = vx / nrm;
    vec2_o[b * 2 + 1] = vy / nrm;
  }

  // ---- K and V (20x256) ----
  for (int e = t; e < 5120; e += 256) {
    int l = e >> 8, c = e & 255;
    const float* wrow = wt + (size_t)(b * 20 + l) * 768;
    float k0 = 0, k1 = 0, v0 = 0, v1 = 0;
    for (int i = 0; i < 768; i += 2) {
      float w0 = wrow[i], w1 = wrow[i + 1];
      k0 = fmaf(w0, Wk[(size_t)(i + 0) * 256 + c], k0);
      k1 = fmaf(w1, Wk[(size_t)(i + 1) * 256 + c], k1);
      v0 = fmaf(w0, Wv[(size_t)(i + 0) * 256 + c], v0);
      v1 = fmaf(w1, Wv[(size_t)(i + 1) * 256 + c], v1);
    }
    s_K[l * 256 + c] = k0 + k1;
    s_V[l * 256 + c] = v0 + v1;
  }
  __syncthreads();

  // ---- Gram G = V V^T (20x20) ----
  for (int e = t; e < 400; e += 256) {
    int l1 = e / 20, l2 = e % 20;
    float a0 = 0, a1 = 0;
    for (int c = 0; c < 256; c += 2) {
      a0 = fmaf(s_V[l1 * 256 + c + 0], s_V[l2 * 256 + c + 0], a0);
      a1 = fmaf(s_V[l1 * 256 + c + 1], s_V[l2 * 256 + c + 1], a1);
    }
    Gm[b * 400 + e] = a0 + a1;
  }

  // ---- M[c2][l] = sum_o Wq[o][c2] * K[l][o], prescaled by 1/sqrt(C)=1/16 ----
  for (int e = t; e < 5120; e += 256) {
    int c2 = e / 20, l = e % 20;
    float m0 = 0, m1 = 0;
    for (int c = 0; c < 256; c += 2) {
      m0 = fmaf(Wq[(size_t)(c + 0) * 256 + c2], s_K[l * 256 + c + 0], m0);
      m1 = fmaf(Wq[(size_t)(c + 1) * 256 + c2], s_K[l * 256 + c + 1], m1);
    }
    s_M[e] = (m0 + m1) * 0.0625f;
  }
  __syncthreads();

  // ---- pack M into bf16 WMMA B-operand blobs ----
  // B (32x16, 16-bit), k order used here (same as A): element j of lane: N = lane&15, K = j + (lane>=16 ? 16 : 0)
  for (int grp = t; grp < 512; grp += 256) {
    int s  = grp >> 6;
    int t2 = (grp >> 5) & 1;
    int lane = grp & 31;
    int hi = lane >> 4;
    int l = (lane & 15) + 16 * t2;
    size_t base = ((size_t)(b * 16 + s * 2 + t2) * 32 + lane) * 16;
    for (int j = 0; j < 16; ++j) {
      int k = s * 32 + j + (hi ? 16 : 0);
      float v = (l < 20) ? s_M[k * 20 + l] : 0.0f;
      blob[base + j] = f2bf(v);
    }
  }
}

// ---------------------------------------------------------------------------
// Pass B: gates via WMMA. One wave = one 16-position tile.
// logits(16x32 padded) = bf16(F-tile) x blob(M),  then per-position softmax
// and gate = sqrt(attn G attn^T), atomic max per batch.
// ---------------------------------------------------------------------------
__global__ void gate_kernel(
    const float* __restrict__ F, const unsigned short* __restrict__ blob,
    const float* __restrict__ Gm, const unsigned* __restrict__ mbits,
    float* __restrict__ gate_raw, unsigned* __restrict__ maxg, int HW)
{
  __shared__ float sG[400];
  __shared__ float slog[8 * 16 * 32];
  const int b = blockIdx.y;
  for (int i = threadIdx.x; i < 400; i += 256) sG[i] = Gm[b * 400 + i];
  __syncthreads();

  const int wave = threadIdx.x >> 5;
  const int lane = threadIdx.x & 31;
  const int tile = blockIdx.x * 8 + wave;
  const int p = lane & 15;
  const int hi = lane >> 4;

  const float* Fb = F + (size_t)b * 256 * HW + (size_t)tile * 16 + p;
  const unsigned short* bb = blob + (size_t)b * 8192;

  v8f c0 = {}; v8f c1 = {};
  for (int s = 0; s < 8; ++s) {
    BfPack A;
    // A (16x32, 16-bit), k order used here (same as B): element j of lane: M = lane&15,
    //   K = j + (lane>=16 ? 16 : 0)
#pragma unroll
    for (int j = 0; j < 16; ++j) {
      int k = s * 32 + j + (hi ? 16 : 0);
      A.u[j] = f2bf(Fb[(size_t)k * HW]);
    }
    v16bf B0 = *(const v16bf*)(bb + ((size_t)(s * 2 + 0) * 32 + lane) * 16);
    v16bf B1 = *(const v16bf*)(bb + ((size_t)(s * 2 + 1) * 32 + lane) * 16);
    c0 = __builtin_amdgcn_wmma_f32_16x16x32_bf16(false, A.v, false, B0, (short)0, c0, false, false);
    c1 = __builtin_amdgcn_wmma_f32_16x16x32_bf16(false, A.v, false, B1, (short)0, c1, false, false);
  }

  // D layout: lane<16: N=lane, M=r ; lane>=16: N=lane-16, M=r+8
  float* myl = slog + wave * 512;
#pragma unroll
  for (int r = 0; r < 8; ++r) {
    myl[(hi * 8 + r) * 32 + p]      = c0[r];
    myl[(hi * 8 + r) * 32 + 16 + p] = c1[r];
  }
  __builtin_amdgcn_wave_barrier();

  const unsigned mb = mbits[b];
  float gr = 0.0f;
  if (lane < 16) {
    const float* lg = myl + lane * 32;
    // load all 20 logits unconditionally (contiguous, 128B-aligned LDS)
    float e[20];
#pragma unroll
    for (int l = 0; l < 20; ++l) e[l] = lg[l];
    // apply mask with pure value selects (no predicated loads)
    float m = -3.0e38f;
#pragma unroll
    for (int l = 0; l < 20; ++l) {
      float v = ((mb >> l) & 1u) ? e[l] : -3.0e38f;
      e[l] = v;
      m = fmaxf(m, v);
    }
    float ssum = 0.0f;
#pragma unroll
    for (int l = 0; l < 20; ++l) {
      float x = __expf(e[l] - m);
      x = ((mb >> l) & 1u) ? x : 0.0f;
      e[l] = x;
      ssum += x;
    }
    float inv = 1.0f / ssum;
    float q = 0.0f;
#pragma unroll
    for (int l1 = 0; l1 < 20; ++l1) {
      const float* Gr = sG + l1 * 20;
      float acc = 0.0f;
#pragma unroll
      for (int l2 = 0; l2 < 20; ++l2) acc = fmaf(Gr[l2], e[l2], acc);
      q = fmaf(e[l1], acc, q);
    }
    q *= inv * inv;
    gr = sqrtf(fmaxf(q, 0.0f));
    gate_raw[(size_t)b * HW + (size_t)tile * 16 + lane] = gr;
  }

  // wave max reduction (lanes >= 16 contribute 0), one atomic per wave
  float v = gr;
  v = fmaxf(v, __shfl_xor(v, 1, 32));
  v = fmaxf(v, __shfl_xor(v, 2, 32));
  v = fmaxf(v, __shfl_xor(v, 4, 32));
  v = fmaxf(v, __shfl_xor(v, 8, 32));
  v = fmaxf(v, __shfl_xor(v, 16, 32));
  if (lane == 0) atomicMax(maxg + b, __float_as_uint(v));   // valid: all values >= 0
}

// ---------------------------------------------------------------------------
// Pass C: in-place gate finalize: g = raw/(max+eps) * sigmoid(direction prior)
// HW = 1<<hwShift, W = 1<<wShift (power-of-two index math, no divisions)
// ---------------------------------------------------------------------------
__global__ void gate_final_kernel(
    float* __restrict__ gates, const unsigned* __restrict__ maxg,
    const float* __restrict__ vec2, int hwShift, int wShift)
{
  int idx = blockIdx.x * 256 + threadIdx.x;
  int b = idx >> hwShift;
  int n = idx & ((1 << hwShift) - 1);
  int h = n >> wShift;
  int w = n & ((1 << wShift) - 1);
  float mx = __uint_as_float(maxg[b]);
  int W = 1 << wShift;
  float step = 2.0f / (float)(W - 1);       // H == W for every scale
  float xs = -1.0f + step * (float)w;
  float ys = -1.0f + step * (float)h;
  float proj = vec2[b * 2 + 0] * xs + vec2[b * 2 + 1] * ys;
  float sig = 1.0f / (1.0f + __expf(-proj));
  gates[idx] = gates[idx] / (mx + 1e-6f) * sig;
}

// ---------------------------------------------------------------------------
// Pass D: mod = gate * F * (1+gamma) + beta   (float4 vectorized)
// HW = 1<<hwShift
// ---------------------------------------------------------------------------
__global__ void mod_kernel(
    const float* __restrict__ F, const float* __restrict__ gates,
    const float* __restrict__ g1p, const float* __restrict__ beta,
    float* __restrict__ outp, int hwShift)
{
  size_t i4 = (size_t)blockIdx.x * 256 + threadIdx.x;
  size_t e = i4 * 4;
  int n = (int)(e & (size_t)((1 << hwShift) - 1));
  size_t bc = e >> hwShift;
  int c = (int)(bc & 255);
  int b = (int)(bc >> 8);
  const float4 f = *(const float4*)(F + e);
  const float4 g = *(const float4*)(gates + ((size_t)b << hwShift) + n);
  float a  = g1p[b * 256 + c];
  float bt = beta[b * 256 + c];
  float4 o;
  o.x = g.x * f.x * a + bt;
  o.y = g.y * f.y * a + bt;
  o.z = g.z * f.z * a + bt;
  o.w = g.w * f.w * a + bt;
  *(float4*)(outp + e) = o;
}

// ---------------------------------------------------------------------------
extern "C" void kernel_launch(void* const* d_in, const int* in_sizes, int n_in,
                              void* d_out, int out_size, void* d_ws, size_t ws_size,
                              hipStream_t stream)
{
  (void)in_sizes; (void)n_in; (void)out_size; (void)ws_size;

  const float* feats[4] = { (const float*)d_in[0], (const float*)d_in[1],
                            (const float*)d_in[2], (const float*)d_in[3] };
  const float* wt   = (const float*)d_in[4];
  const float* sv   = (const float*)d_in[5];
  const int*   mask = (const int*)d_in[6];
  const float* Wq   = (const float*)d_in[7];
  const float* Wk   = (const float*)d_in[8];
  const float* Wv   = (const float*)d_in[9];
  const float* gw1  = (const float*)d_in[10];
  const float* gb1  = (const float*)d_in[11];
  const float* gw2  = (const float*)d_in[12];
  const float* gb2  = (const float*)d_in[13];
  const float* bw1  = (const float*)d_in[14];
  const float* bb1  = (const float*)d_in[15];
  const float* bw2  = (const float*)d_in[16];
  const float* bb2  = (const float*)d_in[17];
  const float* dw1  = (const float*)d_in[18];
  const float* db1  = (const float*)d_in[19];
  const float* dw2  = (const float*)d_in[20];
  const float* db2  = (const float*)d_in[21];

  static const int HWs[4]      = { 16384, 4096, 1024, 256 };
  static const int hwShift[4]  = { 14, 12, 10, 8 };
  static const int wShift[4]   = { 7, 6, 5, 4 };
  // d_out float offsets: mods then gates
  static const size_t modOff[4]  = { 0, 67108864, 83886080, 88080384 };
  static const size_t gateOff[4] = { 89128960, 89391104, 89456640, 89473024 };

  // workspace layout (bytes)
  char* ws = (char*)d_ws;
  unsigned short* blob = (unsigned short*)(ws + 0);        // 262144 B
  float*    Gm    = (float*)(ws + 262144);                 // 25600 B
  float*    g1p   = (float*)(ws + 287744);                 // 16384 B
  float*    beta  = (float*)(ws + 304128);                 // 16384 B
  float*    vec2  = (float*)(ws + 320512);                 // 128 B
  unsigned* mbits = (unsigned*)(ws + 320640);              // 64 B
  unsigned* maxg  = (unsigned*)(ws + 320704);              // 256 B (4 scales x 16 batches)

  float* out = (float*)d_out;

  precompute_kernel<<<16, 256, 0, stream>>>(
      wt, sv, mask, Wq, Wk, Wv,
      gw1, gb1, gw2, gb2, bw1, bb1, bw2, bb2, dw1, db1, dw2, db2,
      blob, Gm, g1p, beta, vec2, mbits, (float*)maxg);

  for (int s = 0; s < 4; ++s) {
    dim3 grid(HWs[s] / 128, 16);
    gate_kernel<<<grid, 256, 0, stream>>>(
        feats[s], blob, Gm, mbits, out + gateOff[s], maxg + s * 16, HWs[s]);
  }
  for (int s = 0; s < 4; ++s) {
    int blocks = (16 * HWs[s]) / 256;
    gate_final_kernel<<<blocks, 256, 0, stream>>>(
        out + gateOff[s], maxg + s * 16, vec2, hwShift[s], wShift[s]);
  }
  for (int s = 0; s < 4; ++s) {
    size_t tot4 = (size_t)16 * 256 * HWs[s] / 4;
    mod_kernel<<<(unsigned)(tot4 / 256), 256, 0, stream>>>(
        feats[s], out + gateOff[s], g1p, beta, out + modOff[s], hwShift[s]);
  }
}
